// DMLHead_59803124629821
// MI455X (gfx1250) — compile-verified, matrix operand order checked by reading, not run on hardware
//
#include <hip/hip_runtime.h>
#include <hip/hip_bf16.h>

// ---------------------------------------------------------------------------
// DMLHead fused kernel for MI455X (gfx1250, wave32, WMMA).
//
// Roofline: 17.2 GFLOP of GEMM vs ~470 MB of mandatory HBM traffic
// (read x: 67MB, write 4 outputs: 402MB) => memory bound (~20us at 23.3TB/s).
// Single fused pass: the 131072x512 dot intermediate lives only in LDS;
// prototypes (512x128 f16 = 128KB) stay L2-resident. Math on
// v_wmma_f32_16x16x32_f16 (f32 accumulate over unit-vector f16 inputs).
// Wave tiling: 2 M-blocks x 8 N-blocks per wave so each B fragment feeds two
// WMMAs and P is read from L2 only 2x per block (~512MB total vs 1GB before).
// ---------------------------------------------------------------------------

typedef _Float16 half_t;
typedef __attribute__((ext_vector_type(16))) _Float16 v16h;
typedef __attribute__((ext_vector_type(8)))  _Float16 v8h;
typedef __attribute__((ext_vector_type(8)))  float    v8f;

#define D_CH   128
#define C_CLS  128
#define NEGN   3
#define HW     16384          // 128*128
#define MP     64             // pixels per workgroup
#define LDA    136            // half stride of A tile rows (16B-aligned, bank-staggered)
#define LDN    517            // f32 stride of dot rows (517 mod 64 = 5 -> conflict-free)
#define EPSF   1e-12f
#define INV2SIG2 2.0f         // 1/(2*0.5^2)
#define BETA   0.3f

// ---------------------------------------------------------------------------
// Kernel 1: build packed f16 prototype matrix P[512][128].
//   rows   0..127 : l2norm(reps_weight[c])
//   rows 128..511 : row 128 + n*128 + c = l2norm(reps_n[c] @ fc_w.T + fc_b + reps_n[c])
// One block per row, 128 threads (one per channel).
// ---------------------------------------------------------------------------
__global__ __launch_bounds__(128) void dml_protos(
    const float* __restrict__ reps_w,   // [128,128]
    const float* __restrict__ fc_w,     // [384,128]
    const float* __restrict__ fc_b,     // [384]
    half_t* __restrict__ P)             // [512,128]
{
  __shared__ float red[128];
  __shared__ float rn[128];
  const int j = blockIdx.x;
  const int d = threadIdx.x;
  const int c = (j < C_CLS) ? j : ((j - C_CLS) & (C_CLS - 1));
  const int n = (j - C_CLS) >> 7;

  // l2norm of reps_weight[c]
  float w = reps_w[c * D_CH + d];
  red[d] = w * w;
  __syncthreads();
  for (int s = 64; s > 0; s >>= 1) { if (d < s) red[d] += red[d + s]; __syncthreads(); }
  float inv = 1.f / fmaxf(sqrtf(red[0]), EPSF);
  float rval = w * inv;

  if (j < C_CLS) { P[j * D_CH + d] = (half_t)rval; return; }  // uniform per block

  rn[d] = rval;
  __syncthreads();

  // neg_raw[dd=d] = sum_k rn[k]*fc_w[n*128+d][k] + fc_b[n*128+d] + rn[d]
  const float* fr = fc_w + (size_t)(n * D_CH + d) * D_CH;
  float acc = fc_b[n * D_CH + d] + rval;
  #pragma unroll 8
  for (int k = 0; k < D_CH; ++k) acc += rn[k] * fr[k];

  __syncthreads();
  red[d] = acc * acc;
  __syncthreads();
  for (int s = 64; s > 0; s >>= 1) { if (d < s) red[d] += red[d + s]; __syncthreads(); }
  float inv2 = 1.f / fmaxf(sqrtf(red[0]), EPSF);
  P[j * D_CH + d] = (half_t)(acc * inv2);
}

// ---------------------------------------------------------------------------
// Kernel 2: fused normalize + GEMM(64x512x128 via WMMA) + epilogue.
// Block: 256 threads (8 waves). Dynamic LDS:
//   sA    : half [64][136]  = 17408 B   (A tile, pixel-major rows)
//   sDot  : f32  [64][517]  = 132352 B  (dot results)
//   sPart : f32  [256]      = 1024 B    (per-pixel partial reductions)
// ---------------------------------------------------------------------------
__global__ __launch_bounds__(256) void dml_main(
    const float*  __restrict__ x,   // [8,128,128,128]
    const half_t* __restrict__ P,   // [512,128]
    float*        __restrict__ out)
{
  extern __shared__ char smem[];
  half_t* sA   = (half_t*)smem;                       // 17408 B
  float*  sDot = (float*)(smem + 17408);              // 132352 B
  float*  sPart= (float*)(smem + 17408 + 132352);     // 1024 B

  const int t  = threadIdx.x;
  const int p  = t & 63;        // pixel within tile
  const int g  = t >> 6;        // channel group 0..3
  const long pi0 = (long)blockIdx.x * MP;
  const int  b   = (int)(pi0 >> 14);      // pi0 / HW
  const int  hw  = (int)(pi0 & 16383);    // pi0 % HW  (tiles never straddle b)

  // ---- Phase 1: load + per-pixel L2 normalize -> f16 A tile in LDS --------
  const float* xb = x + (long)b * D_CH * HW + hw + p;
  float xv[32];
  float ss = 0.f;
  #pragma unroll
  for (int j = 0; j < 32; ++j) {
    const int d = g + 4 * j;
    float v = xb[(long)d * HW];             // coalesced: lanes sweep p
    xv[j] = v; ss += v * v;
  }
  sPart[g * 64 + p] = ss;
  __syncthreads();
  float tot = sPart[p] + sPart[64 + p] + sPart[128 + p] + sPart[192 + p];
  float rnv = 1.f / fmaxf(sqrtf(tot), EPSF);
  #pragma unroll
  for (int j = 0; j < 32; ++j) {
    const int d = g + 4 * j;
    sA[p * LDA + d] = (half_t)(xv[j] * rnv);
  }
  __syncthreads();

  // ---- Phase 2: WMMA GEMM: dot[64][512] = A[64][128] * P^T ----------------
  // Wave wv handles M-blocks {2*(wv>>2), 2*(wv>>2)+1}, N-blocks (wv&3)+4*i.
  // Each B fragment is consumed by two WMMAs (both M-blocks).
  {
    const int lane   = t & 31;
    const int wv     = t >> 5;
    const int mg     = wv >> 2;            // 0..1
    const int nr     = wv & 3;             // N-block residue
    const int laneLo = lane & 15;
    const int laneHi = lane >> 4;

    // A fragments (16x32 f16 layout): lane half-groups are contiguous K runs:
    //   lanes 0-15 : K[0..7],  K[16..23] ; lanes 16-31: K[8..15], K[24..31]
    // => two 16B ds_load_b128 per fragment.
    v16h afr[2][4];
    #pragma unroll
    for (int m = 0; m < 2; ++m) {
      const half_t* arow = sA + ((mg * 2 + m) * 16 + laneLo) * LDA + laneHi * 8;
      #pragma unroll
      for (int kb = 0; kb < 4; ++kb) {
        union { v16h v; v8h h[2]; } u;
        u.h[0] = *(const v8h*)(arow + kb * 32);
        u.h[1] = *(const v8h*)(arow + kb * 32 + 16);
        afr[m][kb] = u.v;
      }
    }

    for (int i = 0; i < 8; ++i) {
      const int nb = nr + 4 * i;
      const half_t* brow = P + (nb * 16 + laneLo) * D_CH + laneHi * 8;
      v8f acc0 = {};
      v8f acc1 = {};
      #pragma unroll
      for (int kb = 0; kb < 4; ++kb) {
        union { v16h v; v8h h[2]; } u;      // B fragment mirrors A layout (N<->M)
        u.h[0] = *(const v8h*)(brow + kb * 32);
        u.h[1] = *(const v8h*)(brow + kb * 32 + 16);
        acc0 = __builtin_amdgcn_wmma_f32_16x16x32_f16(
            false, afr[0][kb], false, u.v, (short)0, acc0, false, false);
        acc1 = __builtin_amdgcn_wmma_f32_16x16x32_f16(
            false, afr[1][kb], false, u.v, (short)0, acc1, false, false);
      }
      // D layout: VGPR v -> row mblk*16 + v + 8*laneHi, col nb*16 + laneLo
      const int nc  = nb * 16 + laneLo;
      const int mr0 = (mg * 2 + 0) * 16 + laneHi * 8;
      const int mr1 = (mg * 2 + 1) * 16 + laneHi * 8;
      #pragma unroll
      for (int v = 0; v < 8; ++v) sDot[(mr0 + v) * LDN + nc] = acc0[v];
      #pragma unroll
      for (int v = 0; v < 8; ++v) sDot[(mr1 + v) * LDN + nc] = acc1[v];
    }
  }
  __syncthreads();

  // ---- Phase 3: fused epilogue -------------------------------------------
  // thread (p,g) handles classes c = g*32 .. g*32+31 for pixel p.
  const long obase  = (long)b * C_CLS * HW + hw + p;           // + c*HW
  const long obase3 = (long)b * (C_CLS * NEGN) * HW + hw + p;  // + (3c+n)*HW
  float* o_cls  = out;
  float* o_dist = out + 16777216L;
  float* o_dneg = out + 33554432L;
  float* o_pori = out + 83886080L;

  float sum_local = 0.f;
  #pragma unroll 4
  for (int cc = 0; cc < 32; ++cc) {
    const int c = g * 32 + cc;
    float dsum = sDot[p * LDN + c];
    float dsq  = fmaxf(2.f - 2.f * dsum, EPSF);
    float dist = sqrtf(dsq);
    o_dist[obase + (long)c * HW] = dist;
    float spo = __expf(-dsq);          // = sqrt(probs_ori), exactly
    float po  = spo * spo;             // probs_ori = exp(-2*dsq)
    o_pori[obase + (long)c * HW] = po;
    float mn = 3.4e38f;
    #pragma unroll
    for (int n = 0; n < NEGN; ++n) {
      float dn  = sDot[p * LDN + C_CLS + (n << 7) + c];
      float dnq = fmaxf(2.f - 2.f * dn, EPSF);
      float dtn = sqrtf(dnq);
      o_dneg[obase3 + (long)(3 * c + n) * HW] = dtn;
      mn = fminf(mn, dtn);
    }
    float tt = dist + BETA * (2.f - mn);
    float pr = __expf(-INV2SIG2 * tt * tt);
    sum_local += pr;
    sDot[p * LDN + c] = spo * pr;      // slot fully consumed above; reuse
  }
  sPart[g * 64 + p] = sum_local;
  __syncthreads();
  float rsum = 1.f / (sPart[p] + sPart[64 + p] + sPart[128 + p] + sPart[192 + p]);
  #pragma unroll 4
  for (int cc = 0; cc < 32; ++cc) {
    const int c = g * 32 + cc;
    o_cls[obase + (long)c * HW] = sDot[p * LDN + c] * rsum;
  }
}

// ---------------------------------------------------------------------------
extern "C" void kernel_launch(void* const* d_in, const int* in_sizes, int n_in,
                              void* d_out, int out_size, void* d_ws, size_t ws_size,
                              hipStream_t stream) {
  const float* x      = (const float*)d_in[0];
  const float* reps_w = (const float*)d_in[1];
  const float* fc_w   = (const float*)d_in[2];
  const float* fc_b   = (const float*)d_in[3];
  half_t* P = (half_t*)d_ws;                       // 512*128 f16 = 128KB

  dml_protos<<<512, 128, 0, stream>>>(reps_w, fc_w, fc_b, P);

  const size_t smem = 17408 + 132352 + 1024;       // 150784 B (<320KB/WGP)
  dml_main<<<2048, 256, smem, stream>>>(x, P, (float*)d_out);
}
